// DeformConvTranspose2d_11759620457161
// MI455X (gfx1250) — compile-verified, matrix operand order checked by reading, not run on hardware
//
#include <hip/hip_runtime.h>

// ---------------------------------------------------------------------------
// DeformConvTranspose2d forward, fused bf16-WMMA GEMM + bilinear atomic scatter
// B=8, Cin=Cout=256, H=W=64, 3x3 (9 taps), stride 2, pad 1, out_pad 1 -> OH=OW=128
// Block = one 16-pixel tile; taps looped inside; A tile staged via
// global_load_async_to_lds_b128 (ASYNCcnt); A fragments hoisted to VGPRs;
// scatter atomics use uniform-base + 32-bit-offset (saddr) addressing.
// ---------------------------------------------------------------------------

typedef __attribute__((ext_vector_type(8)))  __bf16 v8bf;
typedef __attribute__((ext_vector_type(16))) __bf16 v16bf;
typedef __attribute__((ext_vector_type(8)))  float  v8f;

#define B_   8
#define CIN  256
#define COUT 256
#define H_   64
#define W_   64
#define HW   4096            // H_*W_
#define OHW  128
#define OSP  (OHW * OHW)     // 16384
#define KTAP 9

// ---- init output with broadcast bias (d_out is poisoned by harness) -------
__global__ __launch_bounds__(256) void dct_init_out(const float* __restrict__ bias,
                                                    float* __restrict__ out) {
  size_t i = (size_t)blockIdx.x * 256 + threadIdx.x;   // B*COUT*OSP elements
  out[i] = bias[(i >> 14) & (COUT - 1)];               // 16384 = OSP
}

// ---- pack x (B,Cin,H,W f32) -> xT (B*H*W, Cin) bf16 -----------------------
__global__ __launch_bounds__(256) void dct_pack_x(const float* __restrict__ x,
                                                  __bf16* __restrict__ xT) {
  size_t i = (size_t)blockIdx.x * 256 + threadIdx.x;   // B*CIN*HW elements
  int b  = (int)(i >> 20);                             // / (CIN*HW)
  int c  = (int)((i >> 12) & (CIN - 1));
  int hw = (int)(i & (HW - 1));
  xT[((size_t)b * HW + hw) * CIN + c] = (__bf16)x[i];
}

// ---- pack weight (Cin,Cout,3,3 f32) -> wT (tap, Cout, Cin) bf16 -----------
__global__ __launch_bounds__(256) void dct_pack_w(const float* __restrict__ w,
                                                  __bf16* __restrict__ wT) {
  size_t i = (size_t)blockIdx.x * 256 + threadIdx.x;   // CIN*COUT*9 elements
  int c   = (int)(i / (COUT * KTAP));
  int rem = (int)(i - (size_t)c * (COUT * KTAP));
  int o   = rem / KTAP;
  int k   = rem - o * KTAP;
  wT[((size_t)k * COUT + o) * CIN + c] = (__bf16)w[i];
}

// ---- fused: one 16-pixel tile, all 9 taps, 256 couts ----------------------
__global__ __launch_bounds__(256) void dct_fused(
    const __bf16* __restrict__ xT,      // [B*HW][CIN] bf16
    const __bf16* __restrict__ wT,      // [9][COUT][CIN] bf16
    const float*  __restrict__ offset,  // [B][18][H][W]
    const float*  __restrict__ mask,    // [B][9][H][W]
    float*        __restrict__ out)     // [B][COUT][OHW][OHW]
{
  __shared__ alignas(16) __bf16 sA[16 * CIN];   // 8 KB A tile (tap-invariant)
  __shared__ int   sIdx[KTAP * 16 * 4];         // corner indices (b*COUT*OSP folded in)
  __shared__ float sW[KTAP * 16 * 4];           // mask-modulated bilinear weights

  const int pt  = blockIdx.x;          // pixel tile (16 pixels)
  const int tid = threadIdx.x;
  const int p0  = pt * 16;

  // --- Stage A tile via CDNA5 async copy: 256 lanes x 2 x 16B = 8 KB ------
  {
    unsigned long long gaddr =
        (unsigned long long)(uintptr_t)(xT + (size_t)p0 * CIN) + (unsigned long long)tid * 16ull;
    unsigned int laddr = (unsigned int)(uintptr_t)sA + (unsigned int)tid * 16u;
    asm volatile(
        "global_load_async_to_lds_b128 %0, %1, off\n\t"
        "global_load_async_to_lds_b128 %0, %1, off offset:4096"
        :: "v"(laddr), "v"(gaddr) : "memory");
  }

  // --- Per-(tap,pixel) scatter metadata: threads 0..143 --------------------
  if (tid < KTAP * 16) {
    const int k   = tid >> 4;           // tap 0..8
    const int px  = tid & 15;           // pixel in tile
    const int p   = p0 + px;
    const int b   = p >> 12;            // / HW
    const int rem = p & (HW - 1);
    const int h   = rem >> 6;
    const int w   = rem & (W_ - 1);
    const int ky  = k / 3, kx = k - ky * 3;
    const size_t obase = (((size_t)b * (2 * KTAP) + 2 * k) * HW) + (size_t)h * W_ + w;
    const float offy = offset[obase];          // channel 2k
    const float offx = offset[obase + HW];     // channel 2k+1
    const float msk  = mask[(((size_t)b * KTAP + k) * HW) + (size_t)h * W_ + w];
    const float y  = offy + (float)(h * 2 - 1 + ky);   // base_y = h*STRIDE - PAD
    const float xp = offx + (float)(w * 2 - 1 + kx);
    const float y0 = floorf(y);
    const float x0 = floorf(xp);
#pragma unroll
    for (int j = 0; j < 4; ++j) {
      const float yy = y0 + (float)(j >> 1);
      const float xx = x0 + (float)(j & 1);
      float wg = (1.0f - fabsf(y - yy)) * (1.0f - fabsf(xp - xx));
      const bool valid = (yy >= 0.0f) && (yy <= (float)(OHW - 1)) &&
                         (xx >= 0.0f) && (xx <= (float)(OHW - 1));
      wg = valid ? wg * msk : 0.0f;
      const int iy = (int)fminf(fmaxf(yy, 0.0f), (float)(OHW - 1));
      const int ix = (int)fminf(fmaxf(xx, 0.0f), (float)(OHW - 1));
      sIdx[tid * 4 + j] = b * (COUT * OSP) + iy * OHW + ix;
      sW[tid * 4 + j]   = wg;
    }
  }

  asm volatile("s_wait_asynccnt 0x0" ::: "memory");
  __syncthreads();

  const int wave = tid >> 5;     // 8 wave32s
  const int lane = tid & 31;
  const int ln   = lane & 15;    // row (A) / col (B) index 0..15
  const int lg   = lane >> 4;    // K-chunk group per 16-bit WMMA layout

  // --- Hoist tap-invariant A fragments into VGPRs (8 x v16bf = 64 VGPRs) ---
  const __bf16* Arow = sA + (size_t)ln * CIN;
  v16bf afrag[8];
#pragma unroll
  for (int kk = 0; kk < 8; ++kk) {
    const int kb = kk * 32 + lg * 8;   // lanes 0-15: K{0..7,16..23}; 16-31: {8..15,24..31}
    v8bf a0 = *(const v8bf*)(Arow + kb);
    v8bf a1 = *(const v8bf*)(Arow + kb + 16);
    afrag[kk] = __builtin_shufflevector(a0, a1, 0,1,2,3,4,5,6,7,8,9,10,11,12,13,14,15);
  }

  const int cout0 = wave * 32 + ln;    // wave owns couts [wave*32, wave*32+32)

#pragma unroll 1
  for (int k = 0; k < KTAP; ++k) {
    const int mb = k * 64;             // metadata base for this tap
#pragma unroll
    for (int t = 0; t < 2; ++t) {
      const int cout    = cout0 + t * 16;
      const int coutOff = cout * OSP;            // lane-dependent, tile-invariant
      const __bf16* Brow = wT + ((size_t)k * COUT + cout) * CIN;
      __builtin_prefetch(Brow + CIN, 0, 0);      // global_prefetch_b8 (next cout's row)

      v8f acc = {};
#pragma unroll
      for (int kk = 0; kk < 8; ++kk) {  // K = 256 in steps of 32
        const int kb = kk * 32 + lg * 8;
        v8bf b0 = *(const v8bf*)(Brow + kb);
        v8bf b1 = *(const v8bf*)(Brow + kb + 16);
        v16bf bb = __builtin_shufflevector(b0, b1, 0,1,2,3,4,5,6,7,8,9,10,11,12,13,14,15);
        acc = __builtin_amdgcn_wmma_f32_16x16x32_bf16(
            /*neg_a=*/false, afrag[kk], /*neg_b=*/false, bb,
            /*c_mod=*/(short)0, acc, /*reuse_a=*/false, /*reuse_b=*/false);
      }

      // C layout: lane holds cout N=ln; VGPR r -> pixel M = r + 8*lg.
      // Branch-free scatter; address = uniform SGPR base `out` + 32-bit
      // element offset -> saddr-form global_atomic_add_f32 (one v_add each).
#pragma unroll
      for (int r = 0; r < 8; ++r) {
        const int m = r + lg * 8;
        const float v = acc[r];
#pragma unroll
        for (int j = 0; j < 4; ++j) {
          const int full = sIdx[mb + m * 4 + j] + coutOff;   // < 33.5M, fits i32
          atomicAdd(out + full, v * sW[mb + m * 4 + j]);
        }
      }
    }
  }
}

// ---------------------------------------------------------------------------
extern "C" void kernel_launch(void* const* d_in, const int* in_sizes, int n_in,
                              void* d_out, int out_size, void* d_ws, size_t ws_size,
                              hipStream_t stream) {
  (void)in_sizes; (void)n_in; (void)out_size; (void)ws_size;
  const float* x      = (const float*)d_in[0];
  const float* weight = (const float*)d_in[1];
  const float* offset = (const float*)d_in[2];
  const float* mask   = (const float*)d_in[3];
  const float* bias   = (const float*)d_in[4];
  float* out = (float*)d_out;

  __bf16* xT = (__bf16*)d_ws;                                        // 16.78 MB
  __bf16* wT = (__bf16*)((char*)d_ws +
                         (size_t)B_ * HW * CIN * sizeof(__bf16));    // +1.18 MB

  dct_init_out<<<(B_ * COUT * OSP) / 256, 256, 0, stream>>>(bias, out);
  dct_pack_x  <<<(B_ * CIN * HW) / 256, 256, 0, stream>>>(x, xT);
  dct_pack_w  <<<(CIN * COUT * KTAP) / 256, 256, 0, stream>>>(weight, wT);
  dct_fused   <<<B_ * HW / 16, 256, 0, stream>>>(xT, wT, offset, mask, out);
}
